// DecoderLayer_3332894622125
// MI455X (gfx1250) — matrix operand order repaired and checked by reading, not compiled, not run on hardware
//
#include <hip/hip_runtime.h>
#include <hip/hip_bf16.h>

// ---------------------------------------------------------------------------
// Types for CDNA5 WMMA
// ---------------------------------------------------------------------------
typedef __bf16 bf16_t;
typedef __attribute__((ext_vector_type(16))) __bf16 v16bf;
typedef __attribute__((ext_vector_type(8)))  float  v8f;

union F16x16 { v16bf v; unsigned int u[8]; };

__device__ __forceinline__ float waveMax(float v) {
#pragma unroll
  for (int o = 16; o > 0; o >>= 1) v = fmaxf(v, __shfl_xor(v, o, 32));
  return v;
}
__device__ __forceinline__ float waveSum(float v) {
#pragma unroll
  for (int o = 16; o > 0; o >>= 1) v += __shfl_xor(v, o, 32);
  return v;
}

// Low 32 bits of a generic pointer to __shared__ == LDS byte offset on AMDGPU.
__device__ __forceinline__ unsigned ldsOffsetOf(const void* p) {
  return (unsigned)(unsigned long long)p;
}
// CDNA5 async copy: global -> LDS, 16 bytes per lane, tracked by ASYNCcnt.
__device__ __forceinline__ void asyncCopyB128(unsigned lds_off, const void* gptr) {
  asm volatile("global_load_async_to_lds_b128 %0, %1, off"
               :: "v"(lds_off), "v"(gptr) : "memory");
}
__device__ __forceinline__ void waitAsync0() {
  asm volatile("s_wait_asynccnt 0x0" ::: "memory");
}

// ---------------------------------------------------------------------------
// Generic WMMA GEMM:  C[M,N] = act( scale * A[M,K] . B^T + bias )
//   A: bf16 row-major [M,K], lda
//   B: if !BKN: bf16 row-major [N,K] (nn.Linear weight layout), ldb
//      if  BKN: bf16 row-major [K,N] (e.g. V matrix), ldb
//   C: f32 (Cf) and/or bf16 (Cb), ldc
// Optional head-batching via blockIdx.z: base offsets b*s?b + h*s?h (Hn>0).
// Assumes: M % 128 == 0, N % BN == 0, K % 32 == 0 (true for every call here).
// Double-buffered LDS; A (and weight-layout B) staged with
// global_load_async_to_lds_b128 so the next K-tile overlaps the WMMAs.
// ---------------------------------------------------------------------------
template <int BN, bool BKN>
__global__ __launch_bounds__(256) void gemm_wmma(
    const bf16_t* __restrict__ A, const bf16_t* __restrict__ Bm,
    const float* __restrict__ bias,
    float* __restrict__ Cf, bf16_t* __restrict__ Cb,
    int M, int N, int K, int lda, int ldb, int ldc,
    float scale, int relu, int Hn,
    long sAb, long sAh, long sBb, long sBh, long sCb, long sCh)
{
  constexpr int BM      = 128;
  constexpr int WAVES_N = BN / 64;             // 2 (BN=128) or 1 (BN=64)
  constexpr int WAVES_M = 8 / WAVES_N;         // 4 or 8
  constexpr int WTM     = (BM / WAVES_M) / 16; // 2 or 1
  constexpr int WTN     = 4;
  constexpr int TPR     = BN / 16;             // threads per k-row (KN loader)

  __shared__ unsigned int As[2][BM][17];       // 128 x 32 bf16 (+pad) x2 buffers
  __shared__ unsigned int Bs[2][BN][17];       // [n][k] layout x2 buffers

  const int tid  = threadIdx.x;
  const int wave = tid >> 5;
  const int lane = tid & 31;
  const int wm   = wave / WAVES_N;
  const int wn   = wave % WAVES_N;
  const int m0   = blockIdx.y * BM;
  const int n0   = blockIdx.x * BN;

  long ab = 0, bb = 0, cb = 0;
  if (Hn > 0) {
    const int z = blockIdx.z;
    const int bz = z / Hn, hz = z % Hn;
    ab = (long)bz * sAb + (long)hz * sAh;
    bb = (long)bz * sBb + (long)hz * sBh;
    cb = (long)bz * sCb + (long)hz * sCh;
  }
  const bf16_t* Ap = A + ab;
  const bf16_t* Bp = Bm + bb;

  // Per-thread staging coordinates (A and weight-layout B): row tid/2,
  // 16 contiguous K elems (= 32 bytes = 2 x b128) at column (tid&1)*16.
  const int ar = tid >> 1;
  const int ac = (tid & 1) << 4;

  auto issueA = [&](int buf, int k0) {
    const unsigned l = ldsOffsetOf(&As[buf][ar][ac >> 1]);
    const bf16_t* g = Ap + (long)(m0 + ar) * lda + k0 + ac;
    asyncCopyB128(l, g);
    asyncCopyB128(l + 16, g + 8);
  };
  auto issueBW = [&](int buf, int k0) {   // weight layout [N,K]
    if (tid < BN * 2) {
      const unsigned l = ldsOffsetOf(&Bs[buf][ar][ac >> 1]);
      const bf16_t* g = Bp + (long)(n0 + ar) * ldb + k0 + ac;
      asyncCopyB128(l, g);
      asyncCopyB128(l + 16, g + 8);
    }
  };
  // KN layout ([K,N]) needs a transpose into LDS -> synchronous, software-
  // pipelined through registers.
  int4 bq0, bq1;
  auto loadBK = [&](int k0) {
    if (tid < 32 * TPR) {
      const int kk = tid / TPR;
      const int nn = (tid % TPR) << 4;
      const int4* src = reinterpret_cast<const int4*>(Bp + (long)(k0 + kk) * ldb + n0 + nn);
      bq0 = src[0];
      bq1 = src[1];
    }
  };
  auto storeBK = [&](int buf) {
    if (tid < 32 * TPR) {
      const int kk = tid / TPR;
      const int nn = (tid % TPR) << 4;
      const unsigned short* e0 = reinterpret_cast<const unsigned short*>(&bq0);
      const unsigned short* e1 = reinterpret_cast<const unsigned short*>(&bq1);
#pragma unroll
      for (int i = 0; i < 8; ++i)
        reinterpret_cast<unsigned short*>(&Bs[buf][nn + i][kk >> 1])[kk & 1] = e0[i];
#pragma unroll
      for (int i = 0; i < 8; ++i)
        reinterpret_cast<unsigned short*>(&Bs[buf][nn + 8 + i][kk >> 1])[kk & 1] = e1[i];
    }
  };

  v8f acc[WTM][WTN];
#pragma unroll
  for (int i = 0; i < WTM; ++i)
#pragma unroll
    for (int j = 0; j < WTN; ++j)
#pragma unroll
      for (int e = 0; e < 8; ++e) acc[i][j][e] = 0.0f;

  // ---- pipeline prologue
  int buf = 0;
  issueA(0, 0);
  if (!BKN) issueBW(0, 0);
  else      loadBK(0);

  for (int k0 = 0; k0 < K; k0 += 32) {
    const bool hasNext = (k0 + 32 < K);
    waitAsync0();        // current buffer's async tiles landed in LDS
    __syncthreads();     // visible to / ordered with all waves
    if (!BKN) {
      if (hasNext) { issueA(buf ^ 1, k0 + 32); issueBW(buf ^ 1, k0 + 32); }
    } else {
      storeBK(buf);      // registers -> LDS (transposed)
      if (hasNext) { issueA(buf ^ 1, k0 + 32); loadBK(k0 + 32); }
      __syncthreads();   // B-tile ds stores visible before fragment reads
    }

    // ---- A fragments (ISA 7.12.2: 16-bit A 16x32 layout)
    F16x16 afr[WTM];
#pragma unroll
    for (int i = 0; i < WTM; ++i) {
      const int mrow = wm * (WTM * 16) + i * 16 + (lane & 15);
      const int kb   = (lane < 16) ? 0 : 8;
#pragma unroll
      for (int p = 0; p < 8; ++p) {
        const int kk = ((p < 4) ? 0 : 16) + kb + 2 * (p & 3);
        afr[i].u[p] = As[buf][mrow][kk >> 1];
      }
    }
    // ---- B fragments (32x16: same K order as A fragments)
    F16x16 bfr[WTN];
#pragma unroll
    for (int j = 0; j < WTN; ++j) {
      const int nrow = wn * 64 + j * 16 + (lane & 15);
      const int kb   = (lane < 16) ? 0 : 8;
#pragma unroll
      for (int p = 0; p < 8; ++p)
        bfr[j].u[p] = Bs[buf][nrow][(((p < 4) ? 0 : 16) + kb + 2 * (p & 3)) >> 1];
    }
    // ---- WMMA
#pragma unroll
    for (int i = 0; i < WTM; ++i)
#pragma unroll
      for (int j = 0; j < WTN; ++j)
        acc[i][j] = __builtin_amdgcn_wmma_f32_16x16x32_bf16(
            false, afr[i].v, false, bfr[j].v, (short)0, acc[i][j], false, false);

    buf ^= 1;
  }

  // ---- epilogue: scale, bias, relu, store f32 and/or bf16
#pragma unroll
  for (int i = 0; i < WTM; ++i) {
#pragma unroll
    for (int j = 0; j < WTN; ++j) {
      const int n     = n0 + wn * 64 + j * 16 + (lane & 15);
      const int mbase = m0 + wm * (WTM * 16) + i * 16 + ((lane < 16) ? 0 : 8);
      const float bv  = bias ? bias[n] : 0.0f;
#pragma unroll
      for (int r = 0; r < 8; ++r) {
        float v = acc[i][j][r] * scale + bv;
        if (relu) v = fmaxf(v, 0.0f);
        const long idx = cb + (long)(mbase + r) * ldc + n;
        if (Cf) Cf[idx] = v;
        if (Cb) Cb[idx] = (bf16_t)v;
      }
    }
  }
}

// ---------------------------------------------------------------------------
// Row softmax over S columns, in-place on bf16, optional causal mask.
// One block (256 thr) per row; S == 2048 -> 8 elems per thread.
// ---------------------------------------------------------------------------
__global__ __launch_bounds__(256) void softmax_kernel(bf16_t* __restrict__ P,
                                                      int S, int T, int causal)
{
  const long row   = blockIdx.x;
  const int  t     = (int)(row % T);
  const int  valid = causal ? (t + 1) : S;
  bf16_t* p = P + row * (long)S;
  const int tid = threadIdx.x;

  float r[8];
  float mx = -3.0e38f;
#pragma unroll
  for (int i = 0; i < 8; ++i) {
    const int c = tid + i * 256;
    const float v = (c < valid) ? (float)p[c] : -3.0e38f;
    r[i] = v;
    mx = fmaxf(mx, v);
  }
  __shared__ float red[8];
  __shared__ float bcast;
  mx = waveMax(mx);
  if ((tid & 31) == 0) red[tid >> 5] = mx;
  __syncthreads();
  if (tid == 0) {
    float m = red[0];
    for (int i = 1; i < 8; ++i) m = fmaxf(m, red[i]);
    bcast = m;
  }
  __syncthreads();
  mx = bcast;
  float s = 0.0f;
#pragma unroll
  for (int i = 0; i < 8; ++i) {
    const int c = tid + i * 256;
    if (c < valid) { r[i] = __expf(r[i] - mx); s += r[i]; }
    else           { r[i] = 0.0f; }
  }
  s = waveSum(s);
  __syncthreads();
  if ((tid & 31) == 0) red[tid >> 5] = s;
  __syncthreads();
  if (tid == 0) {
    float m = 0.0f;
    for (int i = 0; i < 8; ++i) m += red[i];
    bcast = 1.0f / m;
  }
  __syncthreads();
  const float inv = bcast;
#pragma unroll
  for (int i = 0; i < 8; ++i) {
    const int c = tid + i * 256;
    p[c] = (bf16_t)(r[i] * inv);
  }
}

// ---------------------------------------------------------------------------
// y = LayerNorm(x + a) * g + b ; writes f32 (outF) and optionally bf16 (outB)
// One block (256 thr) per row, D == 1024 -> 4 elems per thread.
// ---------------------------------------------------------------------------
__global__ __launch_bounds__(256) void add_ln_kernel(
    const float* __restrict__ X, const float* __restrict__ A,
    const float* __restrict__ g, const float* __restrict__ b,
    float* __restrict__ outF, bf16_t* __restrict__ outB, int D)
{
  const long row = blockIdx.x;
  const int  tid = threadIdx.x;
  const float* x = X + row * (long)D;
  const float* a = A + row * (long)D;
  float v[4];
  float s = 0.0f, s2 = 0.0f;
#pragma unroll
  for (int i = 0; i < 4; ++i) {
    const int c = tid + i * 256;
    const float t = x[c] + a[c];
    v[i] = t; s += t; s2 += t * t;
  }
  __shared__ float rs_[8], rq_[8];
  __shared__ float bm, bvr;
  s = waveSum(s); s2 = waveSum(s2);
  if ((tid & 31) == 0) { rs_[tid >> 5] = s; rq_[tid >> 5] = s2; }
  __syncthreads();
  if (tid == 0) {
    float ts = 0.0f, tq = 0.0f;
    for (int i = 0; i < 8; ++i) { ts += rs_[i]; tq += rq_[i]; }
    const float mu  = ts / (float)D;
    const float var = tq / (float)D - mu * mu;
    bm = mu; bvr = rsqrtf(var + 1e-5f);
  }
  __syncthreads();
  const float mu = bm, rstd = bvr;
#pragma unroll
  for (int i = 0; i < 4; ++i) {
    const int c = tid + i * 256;
    const float y = (v[i] - mu) * rstd * g[c] + b[c];
    outF[row * (long)D + c] = y;
    if (outB) outB[row * (long)D + c] = (bf16_t)y;
  }
}

// ---------------------------------------------------------------------------
// f32 -> bf16 cast (n % 4 == 0)
// ---------------------------------------------------------------------------
__global__ void cast_bf16_kernel(const float* __restrict__ s,
                                 bf16_t* __restrict__ d, long n)
{
  long i = ((long)blockIdx.x * blockDim.x + threadIdx.x) * 4;
  const long stride = (long)gridDim.x * blockDim.x * 4;
  for (; i + 3 < n; i += stride) {
    const float4 v = *reinterpret_cast<const float4*>(s + i);
    d[i]     = (bf16_t)v.x;
    d[i + 1] = (bf16_t)v.y;
    d[i + 2] = (bf16_t)v.z;
    d[i + 3] = (bf16_t)v.w;
  }
}

// ---------------------------------------------------------------------------
// Host orchestration
// ---------------------------------------------------------------------------
extern "C" void kernel_launch(void* const* d_in, const int* in_sizes, int n_in,
                              void* d_out, int out_size, void* d_ws, size_t ws_size,
                              hipStream_t stream)
{
  (void)in_sizes; (void)n_in; (void)out_size;
  const long Dm = 1024, Hh = 16, DKh = 64, DFFm = 4096, Bn = 2, Tn = 2048, Sn = 2048;
  const long BT = Bn * Tn, BS = Bn * Sn;

  const float* x     = (const float*)d_in[0];
  const float* enc   = (const float*)d_in[1];
  const float* sa_wq = (const float*)d_in[4];  const float* sa_bq = (const float*)d_in[5];
  const float* sa_wk = (const float*)d_in[6];  const float* sa_bk = (const float*)d_in[7];
  const float* sa_wv = (const float*)d_in[8];  const float* sa_bv = (const float*)d_in[9];
  const float* sa_wo = (const float*)d_in[10]; const float* sa_bo = (const float*)d_in[11];
  const float* ca_wq = (const float*)d_in[12]; const float* ca_bq = (const float*)d_in[13];
  const float* ca_wk = (const float*)d_in[14]; const float* ca_bk = (const float*)d_in[15];
  const float* ca_wv = (const float*)d_in[16]; const float* ca_bv = (const float*)d_in[17];
  const float* ca_wo = (const float*)d_in[18]; const float* ca_bo = (const float*)d_in[19];
  const float* ff_w1 = (const float*)d_in[20]; const float* ff_b1 = (const float*)d_in[21];
  const float* ff_w2 = (const float*)d_in[22]; const float* ff_b2 = (const float*)d_in[23];
  const float* ln1g = (const float*)d_in[24]; const float* ln1b = (const float*)d_in[25];
  const float* ln2g = (const float*)d_in[26]; const float* ln2b = (const float*)d_in[27];
  const float* ln3g = (const float*)d_in[28]; const float* ln3b = (const float*)d_in[29];

  // ---- workspace layout
  char* ws = (char*)d_ws;
  size_t off = 0;
  auto take = [&](size_t bytes) -> char* {
    char* p = ws + off;
    off += bytes;
    off = (off + 255) & ~(size_t)255;
    return p;
  };
  bf16_t* wsaq = (bf16_t*)take(Dm * Dm * 2);
  bf16_t* wsak = (bf16_t*)take(Dm * Dm * 2);
  bf16_t* wsav = (bf16_t*)take(Dm * Dm * 2);
  bf16_t* wsao = (bf16_t*)take(Dm * Dm * 2);
  bf16_t* wcaq = (bf16_t*)take(Dm * Dm * 2);
  bf16_t* wcak = (bf16_t*)take(Dm * Dm * 2);
  bf16_t* wcav = (bf16_t*)take(Dm * Dm * 2);
  bf16_t* wcao = (bf16_t*)take(Dm * Dm * 2);
  bf16_t* wff1 = (bf16_t*)take(DFFm * Dm * 2);
  bf16_t* wff2 = (bf16_t*)take(Dm * DFFm * 2);
  bf16_t* x0bf  = (bf16_t*)take(BT * Dm * 2);
  bf16_t* encbf = (bf16_t*)take(BS * Dm * 2);
  bf16_t* x1bf  = (bf16_t*)take(BT * Dm * 2);
  bf16_t* x2bf  = (bf16_t*)take(BT * Dm * 2);
  bf16_t* qbf   = (bf16_t*)take(BT * Dm * 2);   // reused as attention output
  bf16_t* kbf   = (bf16_t*)take(BS * Dm * 2);
  bf16_t* vbf   = (bf16_t*)take(BS * Dm * 2);
  bf16_t* P     = (bf16_t*)take(Bn * Hh * Tn * Sn * 2);
  float*  aout  = (float*)take(BT * Dm * 4);
  float*  x1f   = (float*)take(BT * Dm * 4);
  float*  x2f   = (float*)take(BT * Dm * 4);
  bf16_t* hbf   = (bf16_t*)take(BT * DFFm * 2);
  if (off > ws_size) return;  // deterministic: depends only on problem sizes
  bf16_t* obf = qbf;

  auto cast = [&](const float* s, bf16_t* d, long n) {
    int blocks = (int)((n / 4 + 255) / 256);
    if (blocks > 8192) blocks = 8192;
    cast_bf16_kernel<<<blocks, 256, 0, stream>>>(s, d, n);
  };
  auto gemm128 = [&](const bf16_t* A, const bf16_t* Bw, const float* bias,
                     float* Cf, bf16_t* Cb, int M, int N, int K,
                     int lda, int ldb, int ldc, float scale, int relu) {
    gemm_wmma<128, false><<<dim3(N / 128, M / 128, 1), 256, 0, stream>>>(
        A, Bw, bias, Cf, Cb, M, N, K, lda, ldb, ldc, scale, relu,
        0, 0, 0, 0, 0, 0, 0);
  };

  // ---- precision conversion (f32 -> bf16) for activations and weights
  cast(x, x0bf, BT * Dm);
  cast(enc, encbf, BS * Dm);
  cast(sa_wq, wsaq, Dm * Dm); cast(sa_wk, wsak, Dm * Dm);
  cast(sa_wv, wsav, Dm * Dm); cast(sa_wo, wsao, Dm * Dm);
  cast(ca_wq, wcaq, Dm * Dm); cast(ca_wk, wcak, Dm * Dm);
  cast(ca_wv, wcav, Dm * Dm); cast(ca_wo, wcao, Dm * Dm);
  cast(ff_w1, wff1, DFFm * Dm);
  cast(ff_w2, wff2, Dm * DFFm);

  const float inv_sqrt_dk = 0.125f;

  // ================= self-attention =================
  gemm128(x0bf, wsaq, sa_bq, nullptr, qbf, (int)BT, (int)Dm, (int)Dm,
          (int)Dm, (int)Dm, (int)Dm, 1.0f, 0);
  gemm128(x0bf, wsak, sa_bk, nullptr, kbf, (int)BT, (int)Dm, (int)Dm,
          (int)Dm, (int)Dm, (int)Dm, 1.0f, 0);
  gemm128(x0bf, wsav, sa_bv, nullptr, vbf, (int)BT, (int)Dm, (int)Dm,
          (int)Dm, (int)Dm, (int)Dm, 1.0f, 0);
  // scores[b,h] = Q[b,h] . K[b,h]^T / sqrt(dk)   -> bf16 P
  gemm_wmma<128, false><<<dim3(Sn / 128, Tn / 128, Bn * Hh), 256, 0, stream>>>(
      qbf, kbf, nullptr, nullptr, P, (int)Tn, (int)Sn, (int)DKh,
      (int)Dm, (int)Dm, (int)Sn, inv_sqrt_dk, 0, (int)Hh,
      Tn * Dm, 64L, Sn * Dm, 64L, Hh * Tn * Sn, Tn * Sn);
  softmax_kernel<<<(unsigned)(Bn * Hh * Tn), 256, 0, stream>>>(P, (int)Sn, (int)Tn, 1);
  // O[b,h] = P[b,h] . V[b,h]   (V is [S, dk] row-major -> K-major B)
  gemm_wmma<64, true><<<dim3(1, Tn / 128, Bn * Hh), 256, 0, stream>>>(
      P, vbf, nullptr, nullptr, obf, (int)Tn, (int)DKh, (int)Sn,
      (int)Sn, (int)Dm, (int)Dm, 1.0f, 0, (int)Hh,
      Hh * Tn * Sn, Tn * Sn, Sn * Dm, 64L, Tn * Dm, 64L);
  gemm128(obf, wsao, sa_bo, aout, nullptr, (int)BT, (int)Dm, (int)Dm,
          (int)Dm, (int)Dm, (int)Dm, 1.0f, 0);
  add_ln_kernel<<<(unsigned)BT, 256, 0, stream>>>(x, aout, ln1g, ln1b, x1f, x1bf, (int)Dm);

  // ================= cross-attention =================
  gemm128(x1bf, wcaq, ca_bq, nullptr, qbf, (int)BT, (int)Dm, (int)Dm,
          (int)Dm, (int)Dm, (int)Dm, 1.0f, 0);
  gemm128(encbf, wcak, ca_bk, nullptr, kbf, (int)BS, (int)Dm, (int)Dm,
          (int)Dm, (int)Dm, (int)Dm, 1.0f, 0);
  gemm128(encbf, wcav, ca_bv, nullptr, vbf, (int)BS, (int)Dm, (int)Dm,
          (int)Dm, (int)Dm, (int)Dm, 1.0f, 0);
  gemm_wmma<128, false><<<dim3(Sn / 128, Tn / 128, Bn * Hh), 256, 0, stream>>>(
      qbf, kbf, nullptr, nullptr, P, (int)Tn, (int)Sn, (int)DKh,
      (int)Dm, (int)Dm, (int)Sn, inv_sqrt_dk, 0, (int)Hh,
      Tn * Dm, 64L, Sn * Dm, 64L, Hh * Tn * Sn, Tn * Sn);
  softmax_kernel<<<(unsigned)(Bn * Hh * Tn), 256, 0, stream>>>(P, (int)Sn, (int)Tn, 0);
  gemm_wmma<64, true><<<dim3(1, Tn / 128, Bn * Hh), 256, 0, stream>>>(
      P, vbf, nullptr, nullptr, obf, (int)Tn, (int)DKh, (int)Sn,
      (int)Sn, (int)Dm, (int)Dm, 1.0f, 0, (int)Hh,
      Hh * Tn * Sn, Tn * Sn, Sn * Dm, 64L, Tn * Dm, 64L);
  gemm128(obf, wcao, ca_bo, aout, nullptr, (int)BT, (int)Dm, (int)Dm,
          (int)Dm, (int)Dm, (int)Dm, 1.0f, 0);
  add_ln_kernel<<<(unsigned)BT, 256, 0, stream>>>(x1f, aout, ln2g, ln2b, x2f, x2bf, (int)Dm);

  // ================= FFN =================
  gemm128(x2bf, wff1, ff_b1, nullptr, hbf, (int)BT, (int)DFFm, (int)Dm,
          (int)Dm, (int)Dm, (int)DFFm, 1.0f, 1);
  gemm128(hbf, wff2, ff_b2, aout, nullptr, (int)BT, (int)Dm, (int)DFFm,
          (int)DFFm, (int)DFFm, (int)Dm, 1.0f, 0);
  add_ln_kernel<<<(unsigned)BT, 256, 0, stream>>>(x2f, aout, ln3g, ln3b,
                                                  (float*)d_out, nullptr, (int)Dm);
}
